// RegionProposalNetwork_26731876450582
// MI455X (gfx1250) — matrix operand order repaired and checked by reading, not compiled, not run on hardware
//
#include <hip/hip_runtime.h>
#include <cstdint>
#include <cstddef>

// ---------------- problem constants (mirror the JAX reference) ----------------
#define C_    256
#define H_    248
#define W_    216
#define B_    4
#define A_    2
#define NGT_  20
#define MPIX  (B_*H_*W_)          // 214272 spatial rows of the implicit GEMM
#define M2_   (H_*W_*A_)          // 107136 anchors per image
#define KNMS  2048
#define NPAD  131072              // 2^17 >= M2_, per-image bitonic length
#define LOG2_NPAD 17
#define STRIDE_F 0.4f
#define SCORE_TH 0.3f
#define NMS_TH_F 0.5f
#define HIGH_TH 0.6f
#define LOW_TH  0.2f
#define HALF_PI 1.5707963267948966f

#define LDA 40                    // padded LDS row stride (elements); 80B, 16B-aligned, 20-bank skew
#define LDH 264                   // head weight LDS stride; 528B, 16B-aligned, 4-bank skew

typedef __attribute__((ext_vector_type(16))) __bf16 bf16x16;
typedef __attribute__((ext_vector_type(8)))  float  f32x8;
typedef unsigned long long u64;

// ---------------- small helpers ----------------
__device__ inline unsigned short f32_to_bf16u(float f) {
  unsigned u = __float_as_uint(f);
  unsigned r = u + 0x7FFFu + ((u >> 16) & 1u);   // round-to-nearest-even
  return (unsigned short)(r >> 16);
}

// Load 16 contiguous bf16 (32B) from LDS as a WMMA fragment: 2x ds_load_b128.
__device__ inline bf16x16 frag_ld32(const unsigned short* p) {
  union { bf16x16 v; uint4 q[2]; } u;
  u.q[0] = *(const uint4*)(p);
  u.q[1] = *(const uint4*)(p + 8);
  return u.v;
}

// A fragment (16x32 bf16): rows are stored in natural K order [0..31],
// so lane-half h reads 16 contiguous elements at element offset h*16.
__device__ inline bf16x16 load_a_frag(const unsigned short* lds, int row0, int lane) {
  return frag_ld32(lds + (row0 + (lane & 15)) * LDA + (lane >> 4) * 16);
}
// B fragment (32x16 bf16): tile staged transposed as [n][k]; lane n, k = 16*half + e.
__device__ inline bf16x16 load_b_frag(const unsigned short* lds, int ld, int col0, int lane) {
  return frag_ld32(lds + (col0 + (lane & 15)) * ld + (lane >> 4) * 16);
}

__device__ inline float bev_iou_ag(float acx, float acy, const float* g) {
  float ax1 = acx - 1.95f, ax2 = acx + 1.95f;
  float ay1 = acy - 0.80f, ay2 = acy + 0.80f;
  float gx1 = g[0] - g[3] * 0.5f, gx2 = g[0] + g[3] * 0.5f;
  float gy1 = g[1] - g[4] * 0.5f, gy2 = g[1] + g[4] * 0.5f;
  float iw = fmaxf(fminf(ax2, gx2) - fmaxf(ax1, gx1), 0.0f);
  float ih = fmaxf(fminf(ay2, gy2) - fmaxf(ay1, gy1), 0.0f);
  float inter = iw * ih;
  float aa = (ax2 - ax1) * (ay2 - ay1);
  float ab = (gx2 - gx1) * (gy2 - gy1);
  return inter / (aa + ab - inter + 1e-8f);
}

// ---------------- layout conversion kernels ----------------
// x (B,C,H,W) f32 -> xb (B,H,W,C) bf16 (GEMM A operand, channels contiguous)
__global__ void rpn_cvt_x(const float* __restrict__ x, unsigned short* __restrict__ xb) {
  long long gid = (long long)blockIdx.x * blockDim.x + threadIdx.x;
  if (gid >= (long long)MPIX * C_) return;
  int c = (int)(gid & (C_ - 1));
  long long rest = gid >> 8;
  int w = (int)(rest % W_); rest /= W_;
  int h = (int)(rest % H_);
  int b = (int)(rest / H_);
  float v = x[(((long long)b * C_ + c) * H_ + h) * W_ + w];
  xb[gid] = f32_to_bf16u(v);
}

// rpn_w (O,I,3,3) f32 -> wb[tap][o][i] bf16 (transposed: K contiguous per out-channel row)
__global__ void rpn_cvt_w(const float* __restrict__ wsrc, unsigned short* __restrict__ wb) {
  int gid = blockIdx.x * blockDim.x + threadIdx.x;
  if (gid >= 9 * C_ * C_) return;
  int i = gid & (C_ - 1);
  int rest = gid >> 8;
  int o = rest & (C_ - 1);
  int tap = rest >> 8;
  wb[gid] = f32_to_bf16u(wsrc[((long long)o * C_ + i) * 9 + tap]);
}

// pack cls_w (2,C) + reg_w (18,C) into hb[o32][i] bf16 (transposed), rows 20..31 zero
__global__ void rpn_build_head_w(const float* __restrict__ cls_w, const float* __restrict__ reg_w,
                                 unsigned short* __restrict__ hb) {
  int gid = blockIdx.x * blockDim.x + threadIdx.x;
  if (gid >= 32 * C_) return;
  int i = gid & (C_ - 1);
  int o = gid >> 8;
  float v = 0.0f;
  if (o < A_) v = cls_w[o * C_ + i];
  else if (o < 2 + A_ * 9) v = reg_w[(o - 2) * C_ + i];
  hb[gid] = f32_to_bf16u(v);
}

// Store a 16-element (32B) K-chunk into LDS in natural K order.
// Chunk kseg=0 covers K 0..15 -> positions {0..15}; kseg=1 covers K 16..31 -> {16..31}.
__device__ inline void stage_a_row(unsigned short* dstRow, int kseg, uint4 p0, uint4 p1) {
  *(uint4*)(dstRow + (kseg ? 16 : 0))  = p0;   // low 8 K of the chunk
  *(uint4*)(dstRow + (kseg ? 24 : 8)) = p1;  // high 8 K of the chunk
}

// ---------------- main 3x3 conv as implicit GEMM on WMMA bf16 ----------------
// tile: 128M x 64N, K loop = 9 taps x 8 slices of 32; 8 waves, each 32x32 (2x2 WMMA)
__launch_bounds__(256)
__global__ void rpn_conv3x3_wmma(const unsigned short* __restrict__ xb,
                                 const unsigned short* __restrict__ wb,
                                 const float* __restrict__ bias,
                                 unsigned short* __restrict__ feat) {
  __shared__ alignas(16) unsigned short At[128 * LDA];
  __shared__ alignas(16) unsigned short Bt[64 * LDA];
  const int tid  = threadIdx.x;
  const int lane = tid & 31;
  const int wv   = tid >> 5;
  const int wm   = wv >> 1;       // 0..3
  const int wn   = wv & 1;        // 0..1
  const int m0   = blockIdx.x * 128;
  const int n0   = blockIdx.y * 64;

  f32x8 acc[2][2];
#pragma unroll
  for (int mi = 0; mi < 2; ++mi)
#pragma unroll
    for (int ni = 0; ni < 2; ++ni)
      acc[mi][ni] = (f32x8){0, 0, 0, 0, 0, 0, 0, 0};

  // A-loader coordinates (2 threads per row, one 16-element K chunk each)
  const int lrow = tid >> 1;
  const int kseg = tid & 1;
  const int m    = m0 + lrow;
  const int bb   = m / (H_ * W_);
  const int hw   = m % (H_ * W_);
  const int h0   = hw / W_;
  const int w0   = hw % W_;
  // B-loader coordinates (4 threads per out-channel row, 8 K each; tile staged as [n][k])
  const int orow = tid >> 2;      // 0..63
  const int oseg = tid & 3;       // 0..3

  for (int tap = 0; tap < 9; ++tap) {
    const int dh = tap / 3 - 1, dw = tap % 3 - 1;
    const int hh = h0 + dh, ww = w0 + dw;
    const bool inb = (hh >= 0) & (hh < H_) & (ww >= 0) & (ww < W_);
    const unsigned short* srcA = xb + (((long long)bb * H_ + hh) * W_ + ww) * C_;
    const unsigned short* srcB = wb + (long long)(tap * C_ + n0 + orow) * C_;

    for (int kb = 0; kb < 8; ++kb) {
      const int k0 = kb * 32;
      // ---- stage A tile (128 x 32 bf16, natural K order) ----
      unsigned short* dstRow = &At[lrow * LDA];
      if (inb) {
        const uint4* s = (const uint4*)(srcA + k0 + kseg * 16);
        uint4 p0 = s[0], p1 = s[1];
        if (kb < 7) __builtin_prefetch(srcA + k0 + 32, 0, 1);   // global_prefetch_b8
        stage_a_row(dstRow, kseg, p0, p1);
      } else {
        const uint4 z = {0u, 0u, 0u, 0u};
        stage_a_row(dstRow, kseg, z, z);
      }
      // ---- stage B tile (64 n-rows x 32 K, transposed layout) ----
      *(uint4*)(&Bt[orow * LDA + oseg * 8]) = *(const uint4*)(srcB + k0 + oseg * 8);
      __syncthreads();

      bf16x16 afr0 = load_a_frag(At, wm * 32,      lane);
      bf16x16 afr1 = load_a_frag(At, wm * 32 + 16, lane);
      bf16x16 bfr0 = load_b_frag(Bt, LDA, wn * 32,      lane);
      bf16x16 bfr1 = load_b_frag(Bt, LDA, wn * 32 + 16, lane);

      acc[0][0] = __builtin_amdgcn_wmma_f32_16x16x32_bf16(false, afr0, false, bfr0, (short)0, acc[0][0], false, false);
      acc[0][1] = __builtin_amdgcn_wmma_f32_16x16x32_bf16(false, afr0, false, bfr1, (short)0, acc[0][1], false, false);
      acc[1][0] = __builtin_amdgcn_wmma_f32_16x16x32_bf16(false, afr1, false, bfr0, (short)0, acc[1][0], false, false);
      acc[1][1] = __builtin_amdgcn_wmma_f32_16x16x32_bf16(false, afr1, false, bfr1, (short)0, acc[1][1], false, false);
      __syncthreads();
    }
  }

  // epilogue: bias + ReLU -> feat bf16 (M,C)  (D layout: VGPR r -> M = r + 8*half, N = lane%16)
  const int half = lane >> 4, n = lane & 15;
#pragma unroll
  for (int mi = 0; mi < 2; ++mi)
#pragma unroll
    for (int ni = 0; ni < 2; ++ni) {
      const int og = n0 + wn * 32 + ni * 16 + n;
      const float bsv = bias[og];
#pragma unroll
      for (int r = 0; r < 8; ++r) {
        const int mg = m0 + wm * 32 + mi * 16 + r + 8 * half;
        float v = fmaxf(acc[mi][ni][r] + bsv, 0.0f);
        if (mg < MPIX) feat[(long long)mg * C_ + og] = f32_to_bf16u(v);
      }
    }
}

// ---------------- fused 1x1 heads: GEMM M x 32, K = 256 ----------------
__launch_bounds__(256)
__global__ void rpn_head_wmma(const unsigned short* __restrict__ feat,
                              const unsigned short* __restrict__ hb,
                              const float* __restrict__ cls_b,
                              const float* __restrict__ reg_b,
                              float* __restrict__ obj, float* __restrict__ reg) {
  __shared__ alignas(16) unsigned short At[128 * LDA];
  __shared__ alignas(16) unsigned short Ht[32 * LDH];   // full 32 x 256 weight block, resident
  const int tid = threadIdx.x, lane = tid & 31, wv = tid >> 5;
  const int m0 = blockIdx.x * 128;

  // stage the whole transposed weight block once: 32 rows x 256 K
  {
    const int row = tid >> 3;          // 0..31
    const int seg = tid & 7;           // 0..7, 32 elements each
#pragma unroll
    for (int j = 0; j < 4; ++j)
      *(uint4*)(&Ht[row * LDH + seg * 32 + j * 8]) =
          *(const uint4*)(hb + row * C_ + seg * 32 + j * 8);
  }

  f32x8 acc[2];
  acc[0] = (f32x8){0, 0, 0, 0, 0, 0, 0, 0};
  acc[1] = (f32x8){0, 0, 0, 0, 0, 0, 0, 0};

  const int lrow = tid >> 1, kseg = tid & 1;
  const unsigned short* src = feat + (long long)(m0 + lrow) * C_;

  for (int kb = 0; kb < 8; ++kb) {
    const int k0 = kb * 32;
    const uint4* s = (const uint4*)(src + k0 + kseg * 16);
    uint4 p0 = s[0], p1 = s[1];
    stage_a_row(&At[lrow * LDA], kseg, p0, p1);
    __syncthreads();
    bf16x16 afr = load_a_frag(At, wv * 16, lane);
    bf16x16 b0 = frag_ld32(&Ht[(0  + (lane & 15)) * LDH + k0 + (lane >> 4) * 16]);
    bf16x16 b1 = frag_ld32(&Ht[(16 + (lane & 15)) * LDH + k0 + (lane >> 4) * 16]);
    acc[0] = __builtin_amdgcn_wmma_f32_16x16x32_bf16(false, afr, false, b0, (short)0, acc[0], false, false);
    acc[1] = __builtin_amdgcn_wmma_f32_16x16x32_bf16(false, afr, false, b1, (short)0, acc[1], false, false);
    __syncthreads();
  }

  const int half = lane >> 4, n = lane & 15;
#pragma unroll
  for (int ni = 0; ni < 2; ++ni) {
    const int col = ni * 16 + n;
#pragma unroll
    for (int r = 0; r < 8; ++r) {
      const int mg = m0 + wv * 16 + r + 8 * half;
      if (mg >= MPIX || col >= 2 + A_ * 9) continue;
      const int b = mg / (H_ * W_);
      const int s = mg % (H_ * W_);
      float v = acc[ni][r];
      if (col < A_) {
        obj[(long long)b * M2_ + s * A_ + col] = v + cls_b[col];
      } else {
        const int cc = col - 2, a = cc / 9, jj = cc % 9;
        reg[((long long)b * M2_ + s * A_ + a) * 9 + jj] = v + reg_b[cc];
      }
    }
  }
}

// ---------------- decode + sigmoid + sort-key packing ----------------
__global__ void rpn_decode(const float* __restrict__ obj, const float* __restrict__ reg,
                           float* __restrict__ prop, float* __restrict__ scores,
                           u64* __restrict__ keys) {
  int gid = blockIdx.x * blockDim.x + threadIdx.x;
  if (gid >= B_ * NPAD) return;
  const int b = gid >> LOG2_NPAD;
  const int t = gid & (NPAD - 1);
  u64 key = 0ull;
  if (t < M2_) {
    const int a = t & 1, s = t >> 1, h = s / W_, w = s % W_;
    const float acx = (w + 0.5f) * STRIDE_F, acy = (h + 0.5f) * STRIDE_F;
    const float ayaw = a ? HALF_PI : 0.0f;
    const float* d = reg + ((long long)b * M2_ + t) * 9;
    float* p = prop + ((long long)b * M2_ + t) * 9;
    p[0] = acx + d[0] * 3.9f;
    p[1] = acy + d[1] * 1.6f;
    p[2] = -1.0f + d[2] * 1.56f;
    p[3] = 3.9f * __expf(d[3]);
    p[4] = 1.6f * __expf(d[4]);
    p[5] = 1.56f * __expf(d[5]);
    p[6] = ayaw + d[6];
    p[7] = d[7];
    p[8] = d[8];
    const float x = obj[(long long)b * M2_ + t];
    const float sc = 1.0f / (1.0f + __expf(-x));
    scores[(long long)b * M2_ + t] = sc;
    key = (sc >= SCORE_TH) ? ((((u64)__float_as_uint(sc)) << 32) | (unsigned)t) : (u64)(unsigned)t;
  }
  keys[(long long)b * NPAD + t] = key;
}

// ---------------- bitonic sort substage (descending on u64 keys) ----------------
__global__ void rpn_bitonic(u64* __restrict__ keys, int ksize, int j) {
  int i = blockIdx.x * blockDim.x + threadIdx.x;
  if (i >= B_ * NPAD) return;
  const int b = i >> LOG2_NPAD;
  const int t = i & (NPAD - 1);
  const int p = t ^ j;
  if (p > t) {
    u64* base = keys + (long long)b * NPAD;
    u64 x = base[t], y = base[p];
    const bool descend = ((t & ksize) == 0);
    const bool sw = descend ? (x < y) : (x > y);
    if (sw) { base[t] = y; base[p] = x; }
  }
}

// ---------------- greedy NMS, one workgroup per image ----------------
__launch_bounds__(1024)
__global__ void rpn_nms(const u64* __restrict__ keys, const float* __restrict__ prop,
                        float* __restrict__ out_boxes, float* __restrict__ out_scores,
                        float* __restrict__ out_keep) {
  __shared__ float sx1[KNMS], sy1[KNMS], sx2[KNMS], sy2[KNMS], ssc[KNMS];
  __shared__ int skeep[KNMS];
  const int b = blockIdx.x;
  for (int t = threadIdx.x; t < KNMS; t += 1024) {
    const u64 key = keys[(long long)b * NPAD + t];
    const unsigned fb = (unsigned)(key >> 32);
    const int idx = (int)(key & 0xffffffffu);
    const float* bx = prop + ((long long)b * M2_ + idx) * 9;
    float* ob = out_boxes + ((long long)b * KNMS + t) * 9;
#pragma unroll
    for (int j = 0; j < 9; ++j) ob[j] = bx[j];
    sx1[t] = bx[0] - bx[3] * 0.5f; sx2[t] = bx[0] + bx[3] * 0.5f;
    sy1[t] = bx[1] - bx[4] * 0.5f; sy2[t] = bx[1] + bx[4] * 0.5f;
    ssc[t] = __uint_as_float(fb);
    skeep[t] = (fb != 0u) ? 1 : 0;
  }
  __syncthreads();
  for (int i = 0; i < KNMS; ++i) {
    if (skeep[i]) {
      const float ix1 = sx1[i], iy1 = sy1[i], ix2 = sx2[i], iy2 = sy2[i];
      const float ia = (ix2 - ix1) * (iy2 - iy1);
      for (int jj = threadIdx.x; jj < KNMS; jj += 1024) {
        if (jj > i && skeep[jj]) {
          const float iw = fmaxf(fminf(ix2, sx2[jj]) - fmaxf(ix1, sx1[jj]), 0.0f);
          const float ih = fmaxf(fminf(iy2, sy2[jj]) - fmaxf(iy1, sy1[jj]), 0.0f);
          const float inter = iw * ih;
          const float ja = (sx2[jj] - sx1[jj]) * (sy2[jj] - sy1[jj]);
          const float iou = inter / (ia + ja - inter + 1e-8f);
          if (iou > NMS_TH_F) skeep[jj] = 0;
        }
      }
    }
    __syncthreads();
  }
  for (int t = threadIdx.x; t < KNMS; t += 1024) {
    out_scores[(long long)b * KNMS + t] = skeep[t] ? ssc[t] : 0.0f;
    out_keep[(long long)b * KNMS + t] = skeep[t] ? 1.0f : 0.0f;
  }
}

// ---------------- loss path ----------------
__global__ void rpn_init_losses(float* __restrict__ gt_max, float* __restrict__ losses) {
  int t = threadIdx.x;
  if (t < B_ * NGT_) gt_max[t] = 0.0f;
  if (t < 2) losses[t] = 0.0f;
}

__global__ void rpn_gtmax(const float* __restrict__ targets, float* __restrict__ gt_max) {
  int gid = blockIdx.x * blockDim.x + threadIdx.x;
  if (gid >= B_ * M2_) return;
  const int b = gid / M2_, m2 = gid % M2_;
  const int s = m2 >> 1, h = s / W_, w = s % W_;
  const float acx = (w + 0.5f) * STRIDE_F, acy = (h + 0.5f) * STRIDE_F;
  const float* gt = targets + (long long)b * NGT_ * 9;
#pragma unroll 4
  for (int g = 0; g < NGT_; ++g) {
    const float iou = bev_iou_ag(acx, acy, gt + g * 9);
    atomicMax((unsigned*)&gt_max[b * NGT_ + g], __float_as_uint(iou));   // iou >= 0
  }
}

__launch_bounds__(256)
__global__ void rpn_losses(const float* __restrict__ targets, const float* __restrict__ gt_max,
                           const float* __restrict__ obj, const float* __restrict__ reg,
                           float* __restrict__ losses) {
  __shared__ float s_obj[256], s_reg[256];
  const int gid = blockIdx.x * blockDim.x + threadIdx.x;
  float lo = 0.0f, lr = 0.0f;
  if (gid < B_ * M2_) {
    const int b = gid / M2_, m2 = gid % M2_;
    const int a = m2 & 1, s = m2 >> 1, h = s / W_, w = s % W_;
    const float acx = (w + 0.5f) * STRIDE_F, acy = (h + 0.5f) * STRIDE_F;
    const float* gt = targets + (long long)b * NGT_ * 9;
    float ious[NGT_];
    float best = -1.0f; int bidx = 0;
#pragma unroll 4
    for (int g = 0; g < NGT_; ++g) {
      ious[g] = bev_iou_ag(acx, acy, gt + g * 9);
      if (ious[g] > best) { best = ious[g]; bidx = g; }
    }
    int m = (best >= HIGH_TH) ? bidx : ((best < LOW_TH) ? -1 : -2);
    bool lq = false;
    const float* gm = gt_max + b * NGT_;
#pragma unroll 4
    for (int g = 0; g < NGT_; ++g)
      if (ious[g] == gm[g] && gm[g] > 0.0f) lq = true;
    const int midx = lq ? bidx : m;
    const float label = (midx >= 0) ? 1.0f : ((midx == -1) ? 0.0f : -1.0f);
    const float x = obj[(long long)b * M2_ + m2];
    lo = fmaxf(x, 0.0f) - x * label + log1pf(__expf(-fabsf(x)));
    const float* mg = gt + max(midx, 0) * 9;
    const float anc[9] = {acx, acy, -1.0f, 3.9f, 1.6f, 1.56f, a ? HALF_PI : 0.0f, 0.0f, 0.0f};
    const float* rp = reg + ((long long)b * M2_ + m2) * 9;
#pragma unroll
    for (int j = 0; j < 9; ++j) {
      float tgt;
      if (j < 3)      tgt = (mg[j] - anc[j]) / anc[3 + j];
      else if (j < 6) tgt = __logf(mg[j] / anc[j]);
      else            tgt = mg[j] - anc[j];
      const float d = rp[j] - tgt;
      const float ad = fabsf(d);
      lr += (ad < 1.0f) ? 0.5f * d * d : ad - 0.5f;
    }
  }
  s_obj[threadIdx.x] = lo; s_reg[threadIdx.x] = lr;
  __syncthreads();
  for (int st = 128; st > 0; st >>= 1) {
    if (threadIdx.x < st) { s_obj[threadIdx.x] += s_obj[threadIdx.x + st]; s_reg[threadIdx.x] += s_reg[threadIdx.x + st]; }
    __syncthreads();
  }
  if (threadIdx.x == 0) { atomicAdd(&losses[0], s_obj[0]); atomicAdd(&losses[1], s_reg[0]); }
}

// ---------------- workspace layout ----------------
static constexpr size_t alignUp(size_t x) { return (x + 255) & ~size_t(255); }
static constexpr size_t OFF_XB    = 0;
static constexpr size_t OFF_WB    = alignUp(OFF_XB   + (size_t)MPIX * C_ * 2);
static constexpr size_t OFF_HB    = alignUp(OFF_WB   + 9ull * C_ * C_ * 2);
static constexpr size_t OFF_FEAT  = alignUp(OFF_HB   + 32ull * C_ * 2);
static constexpr size_t OFF_OBJ   = alignUp(OFF_FEAT + (size_t)MPIX * C_ * 2);
static constexpr size_t OFF_REG   = alignUp(OFF_OBJ  + (size_t)B_ * M2_ * 4);
static constexpr size_t OFF_PROP  = alignUp(OFF_REG  + (size_t)B_ * M2_ * 9 * 4);
static constexpr size_t OFF_SC    = alignUp(OFF_PROP + (size_t)B_ * M2_ * 9 * 4);
static constexpr size_t OFF_KEYS  = alignUp(OFF_SC   + (size_t)B_ * M2_ * 4);
static constexpr size_t OFF_GTMAX = alignUp(OFF_KEYS + (size_t)B_ * NPAD * 8);

extern "C" void kernel_launch(void* const* d_in, const int* in_sizes, int n_in,
                              void* d_out, int out_size, void* d_ws, size_t ws_size,
                              hipStream_t stream) {
  (void)in_sizes; (void)n_in; (void)out_size; (void)ws_size;
  const float* x      = (const float*)d_in[0];
  const float* tgts   = (const float*)d_in[1];
  const float* rpn_w  = (const float*)d_in[2];
  const float* rpn_b  = (const float*)d_in[3];
  const float* cls_w  = (const float*)d_in[4];
  const float* cls_b  = (const float*)d_in[5];
  const float* reg_w  = (const float*)d_in[6];
  const float* reg_b  = (const float*)d_in[7];

  char* ws = (char*)d_ws;
  unsigned short* xb   = (unsigned short*)(ws + OFF_XB);
  unsigned short* wb   = (unsigned short*)(ws + OFF_WB);
  unsigned short* hb   = (unsigned short*)(ws + OFF_HB);
  unsigned short* feat = (unsigned short*)(ws + OFF_FEAT);
  float* obj   = (float*)(ws + OFF_OBJ);
  float* reg   = (float*)(ws + OFF_REG);
  float* prop  = (float*)(ws + OFF_PROP);
  float* sc    = (float*)(ws + OFF_SC);
  u64*   keys  = (u64*)(ws + OFF_KEYS);
  float* gtmax = (float*)(ws + OFF_GTMAX);

  float* out        = (float*)d_out;
  float* out_boxes  = out;
  float* out_scores = out + (size_t)B_ * KNMS * 9;
  float* out_keep   = out_scores + (size_t)B_ * KNMS;
  float* out_losses = out_keep + (size_t)B_ * KNMS;

  // 1) precision conversion / repacking (bf16, GEMM-friendly layouts)
  {
    long long n = (long long)MPIX * C_;
    rpn_cvt_x<<<(unsigned)((n + 255) / 256), 256, 0, stream>>>(x, xb);
    rpn_cvt_w<<<(9 * C_ * C_ + 255) / 256, 256, 0, stream>>>(rpn_w, wb);
    rpn_build_head_w<<<(32 * C_ + 255) / 256, 256, 0, stream>>>(cls_w, reg_w, hb);
  }
  // 2) 3x3 conv + ReLU (WMMA bf16 implicit GEMM)
  rpn_conv3x3_wmma<<<dim3(MPIX / 128, C_ / 64), 256, 0, stream>>>(xb, wb, rpn_b, feat);
  // 3) fused cls/reg heads (WMMA, N padded to 32)
  rpn_head_wmma<<<MPIX / 128, 256, 0, stream>>>(feat, hb, cls_b, reg_b, obj, reg);
  // 4) decode + sigmoid + key packing
  rpn_decode<<<(B_ * NPAD + 255) / 256, 256, 0, stream>>>(obj, reg, prop, sc, keys);
  // 5) per-image descending bitonic sort (top-K selection)
  for (int ksz = 2; ksz <= NPAD; ksz <<= 1)
    for (int j = ksz >> 1; j > 0; j >>= 1)
      rpn_bitonic<<<(B_ * NPAD + 255) / 256, 256, 0, stream>>>(keys, ksz, j);
  // 6) greedy NMS + detection outputs
  rpn_nms<<<B_, 1024, 0, stream>>>(keys, prop, out_boxes, out_scores, out_keep);
  // 7) matcher + losses
  rpn_init_losses<<<1, 128, 0, stream>>>(gtmax, out_losses);
  rpn_gtmax<<<(B_ * M2_ + 255) / 256, 256, 0, stream>>>(tgts, gtmax);
  rpn_losses<<<(B_ * M2_ + 255) / 256, 256, 0, stream>>>(tgts, gtmax, obj, reg, out_losses);
}
